// LSTMAE_56788057588490
// MI455X (gfx1250) — matrix operand order repaired and checked by reading, not compiled, not run on hardware
//
#include <hip/hip_runtime.h>
#include <math.h>

#define S_LEN 16384
#define NWG   64       // persistent workgroups per recurrence (each owns H/NWG hidden units)
#define TPB   256      // 8 waves of 32

typedef __attribute__((ext_vector_type(16))) __bf16         v16bf;
typedef __attribute__((ext_vector_type(8)))  float          v8f;
typedef __attribute__((ext_vector_type(8)))  unsigned short us8;

union Frag { struct { us8 lo, hi; } h; v16bf v; };

__device__ __forceinline__ unsigned short f2bf(float f) {
  unsigned u = __builtin_bit_cast(unsigned, f);
  u += 0x7FFFu + ((u >> 16) & 1u);           // round-to-nearest-even
  return (unsigned short)(u >> 16);
}
__device__ __forceinline__ float bf2f(unsigned short b) {
  return __builtin_bit_cast(float, ((unsigned)b) << 16);
}

// ---------------------------------------------------------------------------
// Tensor Data Mover: stage a 2D bf16 tile (rows x cols, row stride in elems)
// from global memory into LDS. Descriptor layout per CDNA5 ISA ch.8 (D#).
// Dynamic LDS starts at offset 0 in this TU (no static LDS), so LDS byte
// offsets are passed numerically.
// ---------------------------------------------------------------------------
#if defined(__has_builtin)
#if __has_builtin(__builtin_amdgcn_tensor_load_to_lds) && __has_builtin(__builtin_amdgcn_s_wait_tensorcnt)
#define HAVE_TDM 1
#endif
#endif

#ifdef HAVE_TDM
typedef __attribute__((ext_vector_type(4))) unsigned int u32x4;
typedef __attribute__((ext_vector_type(4))) int          i32x4;
typedef __attribute__((ext_vector_type(8))) int          i32x8;

__device__ __forceinline__ void tdm_load_2d_bf16(const unsigned short* gsrc,
                                                 unsigned lds_byte_off,
                                                 unsigned rows, unsigned cols,
                                                 unsigned row_stride_elems) {
  unsigned long long ga = (unsigned long long)gsrc;
  u32x4 g0;
  g0[0] = 1u;                                                   // count=1, is_restore=0, gather off
  g0[1] = lds_byte_off;                                         // lds_addr
  g0[2] = (unsigned)(ga & 0xFFFFFFFFull);                       // global_addr[31:0]
  g0[3] = (unsigned)((ga >> 32) & 0x01FFFFFFull) | (2u << 30);  // global_addr[56:32] | type=2
  i32x8 g1;
  g1[0] = (int)(1u << 16);                                      // wg_mask=0, data_size=1 (2 bytes)
  g1[1] = (int)((cols & 0xFFFFu) << 16);                        // tensor_dim0[15:0]
  g1[2] = (int)(((cols >> 16) & 0xFFFFu) | ((rows & 0xFFFFu) << 16)); // dim0 hi | tensor_dim1 lo
  g1[3] = (int)(((rows >> 16) & 0xFFFFu) | ((cols & 0xFFFFu) << 16)); // dim1 hi | tile_dim0
  g1[4] = (int)(rows & 0xFFFFu);                                // tile_dim1 (tile_dim2=0)
  g1[5] = (int)row_stride_elems;                                // tensor_dim0_stride[31:0]
  g1[6] = 0;
  g1[7] = 0;
  i32x4 gz = {0, 0, 0, 0};
#if __clang_major__ >= 23
  i32x8 gz8 = {0, 0, 0, 0, 0, 0, 0, 0};
  __builtin_amdgcn_tensor_load_to_lds(g0, g1, gz, gz, gz8, 0);
#else
  __builtin_amdgcn_tensor_load_to_lds(g0, g1, gz, gz, 0);
#endif
}
#endif

// ---------------------------------------------------------------------------
// fp32 -> bf16 conversion (one-time weight/activation repack)
// ---------------------------------------------------------------------------
__global__ __launch_bounds__(TPB) void cvt_f32_bf16(const float* __restrict__ in,
                                                    unsigned short* __restrict__ out,
                                                    long n) {
  long i = (long)blockIdx.x * TPB + threadIdx.x;
  long stride = (long)gridDim.x * TPB;
  for (; i < n; i += stride) out[i] = f2bf(in[i]);
}

// ---------------------------------------------------------------------------
// WMMA bf16 GEMM:  out[M,N] = A[M,K] @ W[N,K]^T + bias0 + bias1
// Each wave computes a 32x32 tile (2x2 of 16x16), K-loop in steps of 32.
// A fragment layout (ISA 7.12.2, 16-bit A 16x32):
//   lane<16: row=lane,    elems 0-7 -> K=0..7,  elems 8-15 -> K=16..23
//   lane>=16: row=lane-16, elems 0-7 -> K=8..15, elems 8-15 -> K=24..31
// B fragment layout (col = lane&15): same K order as A (lane<16: K=0..7,16..23; lane>=16: K=8..15,24..31)
// ---------------------------------------------------------------------------
__device__ __forceinline__ v16bf loadA(const unsigned short* base, int K, int lane) {
  int r    = lane & 15;
  int koff = (lane < 16) ? 0 : 8;
  const unsigned short* p = base + (long)r * K + koff;
  Frag f;
  f.h.lo = *(const us8*)(p);
  f.h.hi = *(const us8*)(p + 16);
  return f.v;
}
__device__ __forceinline__ v16bf loadB(const unsigned short* base, int K, int lane) {
  int n  = lane & 15;
  int kb = (lane < 16) ? 0 : 8;
  const unsigned short* p = base + (long)n * K + kb;
  Frag f;
  f.h.lo = *(const us8*)(p);
  f.h.hi = *(const us8*)(p + 16);
  return f.v;
}

template <bool OUT_BF16>
__global__ __launch_bounds__(TPB)
void gemm_wmma_bf16(const unsigned short* __restrict__ A,
                    const unsigned short* __restrict__ W,
                    const float* __restrict__ bias0,
                    const float* __restrict__ bias1,
                    void* __restrict__ out,
                    int M, int N, int K) {
  int wid  = (blockIdx.x * TPB + threadIdx.x) >> 5;
  int lane = threadIdx.x & 31;
  int tn32 = N >> 5;
  int tm   = wid / tn32;
  int tn   = wid - tm * tn32;
  if (tm >= (M >> 5)) return;

  const unsigned short* A0 = A + (long)(tm * 32) * K;
  const unsigned short* A1 = A0 + (long)16 * K;
  const unsigned short* W0 = W + (long)(tn * 32) * K;
  const unsigned short* W1 = W0 + (long)16 * K;

  v8f acc00 = {}, acc01 = {}, acc10 = {}, acc11 = {};
  for (int k = 0; k < K; k += 32) {
    if (k + 64 < K) {  // warm L2/L0 for the next K-slab
      __builtin_prefetch(A0 + k + 64, 0, 1);
      __builtin_prefetch(A1 + k + 64, 0, 1);
    }
    v16bf a0 = loadA(A0 + k, K, lane);
    v16bf a1 = loadA(A1 + k, K, lane);
    v16bf b0 = loadB(W0 + k, K, lane);
    v16bf b1 = loadB(W1 + k, K, lane);
    acc00 = __builtin_amdgcn_wmma_f32_16x16x32_bf16(false, a0, false, b0, (short)0, acc00, false, false);
    acc01 = __builtin_amdgcn_wmma_f32_16x16x32_bf16(false, a0, false, b1, (short)0, acc01, false, false);
    acc10 = __builtin_amdgcn_wmma_f32_16x16x32_bf16(false, a1, false, b0, (short)0, acc10, false, false);
    acc11 = __builtin_amdgcn_wmma_f32_16x16x32_bf16(false, a1, false, b1, (short)0, acc11, false, false);
  }

  // C/D layout: VGPR v, lanes 0-15 -> M=v, lanes 16-31 -> M=v+8; col = lane&15
  int rophase = (lane < 16) ? 0 : 8;
  int ncol    = lane & 15;
#pragma unroll
  for (int mi = 0; mi < 2; mi++) {
#pragma unroll
    for (int ni = 0; ni < 2; ni++) {
      v8f acc = (mi == 0) ? (ni == 0 ? acc00 : acc01) : (ni == 0 ? acc10 : acc11);
      int col  = tn * 32 + ni * 16 + ncol;
      float bv = (bias0 ? bias0[col] : 0.f) + (bias1 ? bias1[col] : 0.f);
      int rbase = tm * 32 + mi * 16 + rophase;
#pragma unroll
      for (int v = 0; v < 8; v++) {
        float val = acc[v] + bv;
        long idx  = (long)(rbase + v) * N + col;
        if (OUT_BF16) ((unsigned short*)out)[idx] = f2bf(val);
        else          ((float*)out)[idx]          = val;
      }
    }
  }
}

// ---------------------------------------------------------------------------
// Grid-wide barrier for the persistent recurrence (one per time step).
// ---------------------------------------------------------------------------
__device__ __forceinline__ void grid_barrier(unsigned* cnt, volatile unsigned* gen, unsigned nwg) {
  __syncthreads();
  if (threadIdx.x == 0) {
    __threadfence();
    unsigned g = *gen;
    if (atomicAdd(cnt, 1u) == nwg - 1u) {
      atomicExch(cnt, 0u);
      __threadfence();
      *gen = g + 1u;
    } else {
      while (*gen == g) { __builtin_amdgcn_s_sleep(1); }
    }
    __threadfence();
  }
  __syncthreads();
}

// ---------------------------------------------------------------------------
// Persistent LSTM recurrence. NWG blocks, each owns U = H/NWG hidden units
// (4U gate rows). Whh slice is staged into LDS (bf16) by the Tensor Data
// Mover and stays resident for the whole sequence.
// Input-projection modes:
//   A (x_in != null): fused on-the-fly  xp = Wih(LDS) @ x[t] + b   (IN == 32)
//   B (xp   != null): read precomputed bf16 xp[t] (bias folded by GEMM)
//   C (zvec != null): constant xp = Wz @ z + b, computed once in prologue
// Hidden vector double-buffered in global (hbuf[2H]) -> 1 barrier/step.
// ---------------------------------------------------------------------------
__global__ __launch_bounds__(TPB)
void lstm_persist(const unsigned short* __restrict__ whh,   // [4H x H] bf16
                  const unsigned short* __restrict__ xp,    // [S x 4H] bf16 or null
                  const float* __restrict__ x_in,           // [S x IN] fp32 or null
                  const float* __restrict__ wih,            // [4H x IN] fp32 (mode A)
                  const float* __restrict__ bih,
                  const float* __restrict__ bhh,
                  const float* __restrict__ zvec,           // [HZ] fp32 (mode C)
                  const float* __restrict__ wz,             // [4H x HZ] fp32 (mode C)
                  float* __restrict__ hbuf,                 // [2H] fp32, pre-zeroed
                  unsigned short* __restrict__ hseq,        // [S x H] bf16 or null
                  unsigned* bar_cnt, unsigned* bar_gen,
                  int H, int S, int IN, int HZ) {
  extern __shared__ char smem[];
  const int U = H / NWG;
  const int R = 4 * U;
  unsigned short* whh_s = (unsigned short*)smem;                 // R*H bf16 (at LDS offset 0)
  float* h_s   = (float*)(smem + (size_t)R * H * 2);             // H
  float* z_s   = h_s + H;                                        // R (gate pre-acts)
  float* c_s   = z_s + R;                                        // U (cell state)
  float* aux_s = c_s + U;                                        // mode A: Wih[R*IN]+b[R]; mode C: xp[R]

  const int tid  = threadIdx.x;
  const int lane = tid & 31;
  const int wv   = tid >> 5;
  const int blk  = blockIdx.x;
  const bool modeA = (x_in != nullptr);
  const bool modeC = (zvec != nullptr);

  // --- prologue: stage Whh slice into LDS (TDM DMA; 4 gate tiles of U x H) ---
#ifdef HAVE_TDM
  if (tid < 32) {  // one wave issues the DMA descriptors
    for (int g = 0; g < 4; g++) {
      tdm_load_2d_bf16(whh + (long)(g * H + blk * U) * H,
                       (unsigned)((size_t)g * U * H * 2),   // whh_s is at LDS offset 0
                       (unsigned)U, (unsigned)H, (unsigned)H);
    }
    __builtin_amdgcn_s_wait_tensorcnt(0);
  }
#else
  for (int idx = tid; idx < R * H; idx += TPB) {
    int lr = idx / H, k = idx - lr * H;
    int g = lr / U, u = lr - g * U;
    whh_s[idx] = whh[(long)(g * H + blk * U + u) * H + k];
  }
#endif
  if (modeA) {
    float* wih_s = aux_s;
    float* b_s   = aux_s + R * IN;
    for (int idx = tid; idx < R * IN; idx += TPB) {
      int lr = idx / IN, k = idx - lr * IN;
      int g = lr / U, u = lr - g * U;
      wih_s[idx] = wih[(long)(g * H + blk * U + u) * IN + k];
    }
    for (int lr = tid; lr < R; lr += TPB) {
      int g = lr / U, u = lr - g * U;
      int grow = g * H + blk * U + u;
      b_s[lr] = bih[grow] + bhh[grow];
    }
  }
  if (modeC) {
    for (int lr = tid; lr < R; lr += TPB) {
      int g = lr / U, u = lr - g * U;
      int grow = g * H + blk * U + u;
      float acc = bih[grow] + bhh[grow];
      for (int k = 0; k < HZ; k++) acc += wz[(long)grow * HZ + k] * zvec[k];
      aux_s[lr] = acc;
    }
  }
  for (int u = tid; u < U; u += TPB) c_s[u] = 0.f;
  __syncthreads();

  const int RPW = R / 8;  // gate rows per wave
  for (int t = 0; t < S; t++) {
    // stage h[t] (written last step, visible via barrier fences)
    const float* hp = hbuf + (size_t)(t & 1) * H;
    for (int k = tid; k < H; k += TPB) h_s[k] = hp[k];
    __syncthreads();

    float xreg = 0.f;
    if (modeA) xreg = x_in[(long)t * IN + lane];   // IN == 32 == wave width

    for (int rr = 0; rr < RPW; rr++) {
      int lr = wv * RPW + rr;
      const unsigned short* wrow = whh_s + (long)lr * H;
      float p = modeA ? (aux_s[lr * IN + lane] * xreg) : 0.f;
      // 4-wide LDS dot product: ds_load_b64 (4 bf16) + ds_load_b128 (4 f32)
      for (int k = lane * 4; k < H; k += 128) {
        uint2  w4 = *(const uint2*)(wrow + k);
        float4 h4 = *(const float4*)(h_s + k);
        p += bf2f((unsigned short)(w4.x & 0xFFFFu)) * h4.x;
        p += bf2f((unsigned short)(w4.x >> 16))     * h4.y;
        p += bf2f((unsigned short)(w4.y & 0xFFFFu)) * h4.z;
        p += bf2f((unsigned short)(w4.y >> 16))     * h4.w;
      }
#pragma unroll
      for (int off = 16; off > 0; off >>= 1) p += __shfl_xor(p, off, 32);
      if (lane == 0) {
        float xv;
        if (modeA)      xv = aux_s[R * IN + lr];
        else if (modeC) xv = aux_s[lr];
        else {
          int g = lr / U, u = lr - g * U;
          xv = bf2f(xp[(long)t * (4 * H) + g * H + blk * U + u]);
        }
        z_s[lr] = p + xv;
      }
    }
    __syncthreads();

    if (tid < U) {
      int u = tid;
      float zi = z_s[u], zf = z_s[U + u], zg = z_s[2 * U + u], zo = z_s[3 * U + u];
      float ig = 1.f / (1.f + __expf(-zi));
      float fg = 1.f / (1.f + __expf(-zf));
      float og = 1.f / (1.f + __expf(-zo));
      float c  = fg * c_s[u] + ig * tanhf(zg);
      float h  = og * tanhf(c);
      c_s[u] = c;
      int hidx = blk * U + u;
      hbuf[(size_t)((t + 1) & 1) * H + hidx] = h;
      if (hseq) hseq[(long)t * H + hidx] = f2bf(h);
    }
    grid_barrier(bar_cnt, bar_gen, NWG);
  }
}

// ---------------------------------------------------------------------------
// Host orchestration
// ---------------------------------------------------------------------------
extern "C" void kernel_launch(void* const* d_in, const int* in_sizes, int n_in,
                              void* d_out, int out_size, void* d_ws, size_t ws_size,
                              hipStream_t stream) {
  (void)in_sizes; (void)n_in; (void)out_size; (void)ws_size;
  const float* x       = (const float*)d_in[0];
  const float* e1_Wih  = (const float*)d_in[1];
  const float* e1_Whh  = (const float*)d_in[2];
  const float* e1_bih  = (const float*)d_in[3];
  const float* e1_bhh  = (const float*)d_in[4];
  const float* e2_Wih  = (const float*)d_in[5];
  const float* e2_Whh  = (const float*)d_in[6];
  const float* e2_bih  = (const float*)d_in[7];
  const float* e2_bhh  = (const float*)d_in[8];
  const float* d1_Wih  = (const float*)d_in[9];
  const float* d1_Whh  = (const float*)d_in[10];
  const float* d1_bih  = (const float*)d_in[11];
  const float* d1_bhh  = (const float*)d_in[12];
  const float* d2_Wih  = (const float*)d_in[13];
  const float* d2_Whh  = (const float*)d_in[14];
  const float* d2_bih  = (const float*)d_in[15];
  const float* d2_bhh  = (const float*)d_in[16];
  const float* lin_W   = (const float*)d_in[17];
  const float* lin_b   = (const float*)d_in[18];

  size_t off = 0;
  auto alloc = [&](size_t bytes) -> void* {
    void* p = (char*)d_ws + off;
    off = (off + bytes + 255) & ~(size_t)255;
    return p;
  };
  unsigned short* wbE1Whh = (unsigned short*)alloc(4096ul * 1024 * 2);
  unsigned short* wbE2Whh = (unsigned short*)alloc(2048ul * 512 * 2);
  unsigned short* wbD1Whh = (unsigned short*)alloc(2048ul * 512 * 2);
  unsigned short* wbD2Whh = (unsigned short*)alloc(4096ul * 1024 * 2);
  unsigned short* wbE2Wih = (unsigned short*)alloc(2048ul * 1024 * 2);
  unsigned short* wbD2Wih = (unsigned short*)alloc(4096ul * 512 * 2);
  unsigned short* wbLinW  = (unsigned short*)alloc(32ul * 1024 * 2);
  unsigned short* h1seq   = (unsigned short*)alloc((size_t)S_LEN * 1024 * 2);
  unsigned short* h3seq   = (unsigned short*)alloc((size_t)S_LEN * 512 * 2);
  unsigned short* h4seq   = (unsigned short*)alloc((size_t)S_LEN * 1024 * 2);
  unsigned short* xp2     = (unsigned short*)alloc((size_t)S_LEN * 2048 * 2);
  unsigned short* xp4     = (unsigned short*)alloc((size_t)S_LEN * 4096 * 2);
  float* hb1 = (float*)alloc(2 * 1024 * 4);
  float* hb2 = (float*)alloc(2 * 512 * 4);
  float* hb3 = (float*)alloc(2 * 512 * 4);
  float* hb4 = (float*)alloc(2 * 1024 * 4);
  unsigned* bars = (unsigned*)alloc(16 * 4);

  hipMemsetAsync(hb1, 0, 2 * 1024 * 4, stream);
  hipMemsetAsync(hb2, 0, 2 * 512 * 4, stream);
  hipMemsetAsync(hb3, 0, 2 * 512 * 4, stream);
  hipMemsetAsync(hb4, 0, 2 * 1024 * 4, stream);
  hipMemsetAsync(bars, 0, 16 * 4, stream);

  // bf16 weight repacks
  cvt_f32_bf16<<<2048, TPB, 0, stream>>>(e1_Whh, wbE1Whh, 4096L * 1024);
  cvt_f32_bf16<<<1024, TPB, 0, stream>>>(e2_Whh, wbE2Whh, 2048L * 512);
  cvt_f32_bf16<<<1024, TPB, 0, stream>>>(d1_Whh, wbD1Whh, 2048L * 512);
  cvt_f32_bf16<<<2048, TPB, 0, stream>>>(d2_Whh, wbD2Whh, 4096L * 1024);
  cvt_f32_bf16<<<1024, TPB, 0, stream>>>(e2_Wih, wbE2Wih, 2048L * 1024);
  cvt_f32_bf16<<<1024, TPB, 0, stream>>>(d2_Wih, wbD2Wih, 4096L * 512);
  cvt_f32_bf16<<<64,   TPB, 0, stream>>>(lin_W,  wbLinW,  32L * 1024);

  auto lstm_lds = [](int H, int auxFloats) -> size_t {
    int U = H / NWG, R = 4 * U;
    return (size_t)R * H * 2 + (size_t)(H + R + U + auxFloats) * 4;
  };
  auto launch_gemm = [&](const unsigned short* A, const unsigned short* W,
                         const float* b0, const float* b1, void* out,
                         int M, int N, int K, bool obf) {
    long waves = (long)(M / 32) * (N / 32);
    int blocks = (int)((waves + 7) / 8);
    if (obf) gemm_wmma_bf16<true ><<<blocks, TPB, 0, stream>>>(A, W, b0, b1, out, M, N, K);
    else     gemm_wmma_bf16<false><<<blocks, TPB, 0, stream>>>(A, W, b0, b1, out, M, N, K);
  };

  // encoder layer 1 (H=1024): fused 32-wide input projection, Whh LDS-resident
  lstm_persist<<<NWG, TPB, lstm_lds(1024, 64 * 32 + 64), stream>>>(
      wbE1Whh, nullptr, x, e1_Wih, e1_bih, e1_bhh, nullptr, nullptr,
      hb1, h1seq, bars + 0, bars + 1, 1024, S_LEN, 32, 0);

  // xp2 = h1seq @ e2_Wih^T + b   (WMMA, M=16384 N=2048 K=1024)
  launch_gemm(h1seq, wbE2Wih, e2_bih, e2_bhh, xp2, S_LEN, 2048, 1024, true);

  // encoder layer 2 (H=512): consumes precomputed xp2; only final h needed
  lstm_persist<<<NWG, TPB, lstm_lds(512, 0), stream>>>(
      wbE2Whh, xp2, nullptr, nullptr, nullptr, nullptr, nullptr, nullptr,
      hb2, nullptr, bars + 2, bars + 3, 512, S_LEN, 0, 0);

  // decoder layer 1 (H=512): input is constant z = final e2 hidden (hbuf[0], S even)
  lstm_persist<<<NWG, TPB, lstm_lds(512, 32), stream>>>(
      wbD1Whh, nullptr, nullptr, nullptr, d1_bih, d1_bhh, hb2, d1_Wih,
      hb3, h3seq, bars + 4, bars + 5, 512, S_LEN, 0, 512);

  // xp4 = h3seq @ d2_Wih^T + b   (WMMA, M=16384 N=4096 K=512)
  launch_gemm(h3seq, wbD2Wih, d2_bih, d2_bhh, xp4, S_LEN, 4096, 512, true);

  // decoder layer 2 (H=1024)
  lstm_persist<<<NWG, TPB, lstm_lds(1024, 0), stream>>>(
      wbD2Whh, xp4, nullptr, nullptr, nullptr, nullptr, nullptr, nullptr,
      hb4, h4seq, bars + 6, bars + 7, 1024, S_LEN, 0, 0);

  // out = h4seq @ lin_W^T + lin_b  (WMMA, M=16384 N=32 K=1024, fp32 out)
  launch_gemm(h4seq, wbLinW, lin_b, nullptr, d_out, S_LEN, 32, 1024, false);
}
